// Attend_89103391523208
// MI455X (gfx1250) — matrix operand order repaired and checked by reading, not compiled, not run on hardware
//
#include <hip/hip_runtime.h>

typedef __attribute__((ext_vector_type(16))) _Float16 v16h;
typedef __attribute__((ext_vector_type(8)))  _Float16 v8h;
typedef __attribute__((ext_vector_type(2)))  _Float16 h2v;
typedef __attribute__((ext_vector_type(8)))  float    v8f;
typedef __attribute__((ext_vector_type(4)))  float    f4v;

#define B_   4
#define H_   16
#define N_   2048
#define D_   64
#define KB   32                   // keys per block iteration
#define NITER (N_ / KB)           // 64
#define SCL2 0.18033688011112042f // log2(e)/sqrt(D): folds softmax scale + base-2 exp into Q

__device__ __forceinline__ v16h cat8(v8h lo, v8h hi) {
  return __builtin_shufflevector(lo, hi, 0,1,2,3,4,5,6,7,8,9,10,11,12,13,14,15);
}

__global__ __launch_bounds__(128)
void attn_fwd_f16wmma(const float* __restrict__ q,
                      const float* __restrict__ k,
                      const float* __restrict__ v,
                      float* __restrict__ out)
{
  // double-buffered K tile [key][d] and transposed V tile [d][key], f16
  __shared__ __align__(32) _Float16 Klds[2][KB * D_];   // 2 x 4 KB
  __shared__ __align__(32) _Float16 Vt[2][D_ * KB];     // 2 x 4 KB
  // per-wave P tile [q][key]
  __shared__ __align__(32) _Float16 Plds[4][16][KB];    // 4 KB

  const int tid  = threadIdx.x;
  const int wv   = tid >> 5;
  const int lane = tid & 31;
  const int n    = lane & 15;     // query column / A-row index
  const int hh   = lane >> 4;     // lane half

  const int bh    = blockIdx.x >> 5;     // b*H + h
  const int qblk  = blockIdx.x & 31;
  const int b     = bh >> 4;             // H_ == 16
  const int qbase = qblk * 64 + wv * 16;

  // ---- Q fragments (B-operand of S^T = K x Q^T), pre-scaled by log2(e)/sqrt(D) ----
  const float* qrow = q + ((size_t)bh * N_ + (size_t)(qbase + n)) * D_;
  v16h bq[2];
  #pragma unroll
  for (int c = 0; c < 2; ++c) {
    const float* p = qrow + c * 32 + hh * 8;
    #pragma unroll
    for (int i = 0; i < 16; ++i) bq[c][i] = (_Float16)(p[i < 8 ? i : i + 8] * SCL2);
  }

  // cooperative staging maps
  const int crow = tid >> 2;          // K: key row,  cols [16*cg, 16*cg+16)
  const int cg   = tid & 3;
  const int vg   = tid >> 4;          // V: d-group (8 d), key pair
  const int d0   = vg * 8;
  const int k0   = (tid & 15) * 2;
  const float* kbase = k + (size_t)b * N_ * D_ + (size_t)cg * 16;
  const float* vbase = v + (size_t)b * N_ * D_ + (size_t)d0;

  float m_run = -__builtin_inff();
  float l_run = 0.0f;
  v8f acc[4];
  #pragma unroll
  for (int dc = 0; dc < 4; ++dc) acc[dc] = (v8f){0,0,0,0,0,0,0,0};

  float kraw[16], vraw0[8], vraw1[8];

  // ---- prologue: stage tile 0 into buffer 0 ----
  {
    const float* kp = kbase + (size_t)crow * D_;
    #pragma unroll
    for (int i = 0; i < 16; ++i) kraw[i] = kp[i];
    const float* vp = vbase + (size_t)k0 * D_;
    #pragma unroll
    for (int i = 0; i < 8; ++i) { vraw0[i] = vp[i]; vraw1[i] = vp[D_ + i]; }

    v16h kh;
    #pragma unroll
    for (int i = 0; i < 16; ++i) kh[i] = (_Float16)kraw[i];
    *(v16h*)&Klds[0][crow * D_ + cg * 16] = kh;
    #pragma unroll
    for (int i = 0; i < 8; ++i) {
      h2v t; t[0] = (_Float16)vraw0[i]; t[1] = (_Float16)vraw1[i];
      *(h2v*)&Vt[0][(d0 + i) * KB + k0] = t;       // key-pair packed store
    }
  }
  __syncthreads();

  for (int kb = 0; kb < NITER; ++kb) {
    const int cur = kb & 1;

    // ---- issue next tile's global loads (wait sinks past the WMMAs) ----
    if (kb + 1 < NITER) {
      const float* kp = kbase + (size_t)((kb + 1) * KB + crow) * D_;
      #pragma unroll
      for (int i = 0; i < 16; ++i) kraw[i] = kp[i];
      const float* vp = vbase + (size_t)((kb + 1) * KB + k0) * D_;
      #pragma unroll
      for (int i = 0; i < 8; ++i) { vraw0[i] = vp[i]; vraw1[i] = vp[D_ + i]; }
    }
    if (kb + 2 < NITER) {
      __builtin_prefetch(kbase + (size_t)((kb + 2) * KB + crow) * D_, 0, 3);
      __builtin_prefetch(vbase + (size_t)((kb + 2) * KB + k0) * D_, 0, 3);
    }

    // ---- S^T = K x Q^T : two 16-key C tiles, K-dim 64 in two 32-chunks ----
    v8f st0 = (v8f){0,0,0,0,0,0,0,0};
    v8f st1 = (v8f){0,0,0,0,0,0,0,0};
    #pragma unroll
    for (int c = 0; c < 2; ++c) {
      const int dlo = c * 32 + hh * 8;
      v16h aK0 = cat8(*(const v8h*)&Klds[cur][ n       * D_ + dlo],
                      *(const v8h*)&Klds[cur][ n       * D_ + dlo + 16]);
      v16h aK1 = cat8(*(const v8h*)&Klds[cur][(n + 16) * D_ + dlo],
                      *(const v8h*)&Klds[cur][(n + 16) * D_ + dlo + 16]);
      st0 = __builtin_amdgcn_wmma_f32_16x16x32_f16(false, aK0, false, bq[c],
                                                   (short)0, st0, false, false);
      st1 = __builtin_amdgcn_wmma_f32_16x16x32_f16(false, aK1, false, bq[c],
                                                   (short)0, st1, false, false);
    }

    // ---- online softmax in base-2 domain (scores are pre-scaled) ----
    float mx = -__builtin_inff();
    #pragma unroll
    for (int r = 0; r < 8; ++r) mx = fmaxf(mx, fmaxf(st0[r], st1[r]));
    mx = fmaxf(mx, __shfl_xor(mx, 16, 32));
    const float m_new = fmaxf(m_run, mx);
    const float corr  = __builtin_amdgcn_exp2f(m_run - m_new);
    m_run = m_new;

    float lsum = 0.0f;
    v8h pa, pb;
    #pragma unroll
    for (int r = 0; r < 8; ++r) {
      float e0 = __builtin_amdgcn_exp2f(st0[r] - m_new);
      float e1 = __builtin_amdgcn_exp2f(st1[r] - m_new);
      lsum += e0 + e1;
      pa[r] = (_Float16)e0;
      pb[r] = (_Float16)e1;
    }
    lsum += __shfl_xor(lsum, 16, 32);
    l_run = l_run * corr + lsum;

    #pragma unroll
    for (int dc = 0; dc < 4; ++dc)
      #pragma unroll
      for (int r = 0; r < 8; ++r) acc[dc][r] *= corr;

    // ---- P round-trip through per-wave LDS (contiguous 16B packed stores) ----
    *(v8h*)&Plds[wv][n][     8 * hh] = pa;   // keys  8h .. 8h+7
    *(v8h*)&Plds[wv][n][16 + 8 * hh] = pb;   // keys 16+8h .. 16+8h+7
    asm volatile("s_wait_dscnt 0" ::: "memory");   // same-wave DS is in-order
    v16h bp = cat8(*(const v8h*)&Plds[wv][n][8 * hh], *(const v8h*)&Plds[wv][n][16 + 8 * hh]);

    // ---- O^T += V^T x P^T : 4 d-chunks ----
    #pragma unroll
    for (int dc = 0; dc < 4; ++dc) {
      const int dd = dc * 16 + n;
      v16h aV = cat8(*(const v8h*)&Vt[cur][dd * KB +      hh * 8],
                     *(const v8h*)&Vt[cur][dd * KB + 16 + hh * 8]);
      acc[dc] = __builtin_amdgcn_wmma_f32_16x16x32_f16(false, aV, false, bp,
                                                       (short)0, acc[dc],
                                                       false, false);
    }

    // ---- convert + store next tile into the other buffer ----
    if (kb + 1 < NITER) {
      v16h kh;
      #pragma unroll
      for (int i = 0; i < 16; ++i) kh[i] = (_Float16)kraw[i];
      *(v16h*)&Klds[cur ^ 1][crow * D_ + cg * 16] = kh;
      #pragma unroll
      for (int i = 0; i < 8; ++i) {
        h2v t; t[0] = (_Float16)vraw0[i]; t[1] = (_Float16)vraw1[i];
        *(h2v*)&Vt[cur ^ 1][(d0 + i) * KB + k0] = t;
      }
    }
    __syncthreads();   // one barrier per iteration (double-buffered tiles)
  }

  // ---- epilogue: O[q][d] = acc / l ; lane holds fixed q = n ----
  const float inv = 1.0f / l_run;
  float* orow = out + ((size_t)bh * N_ + (size_t)(qbase + n)) * D_;
  #pragma unroll
  for (int dc = 0; dc < 4; ++dc) {
    f4v lo, hi;
    #pragma unroll
    for (int r = 0; r < 4; ++r) {
      lo[r] = acc[dc][r]     * inv;
      hi[r] = acc[dc][4 + r] * inv;
    }
    *(f4v*)&orow[dc * 16 + hh * 8]     = lo;
    *(f4v*)&orow[dc * 16 + hh * 8 + 4] = hi;
  }
}

extern "C" void kernel_launch(void* const* d_in, const int* in_sizes, int n_in,
                              void* d_out, int out_size, void* d_ws, size_t ws_size,
                              hipStream_t stream) {
  const float* q = (const float*)d_in[0];
  const float* k = (const float*)d_in[1];
  const float* v = (const float*)d_in[2];
  float* out = (float*)d_out;

  dim3 grid(B_ * H_ * (N_ / 64));   // 2048 blocks: (b,h) x 64-query blocks
  dim3 block(128);                  // 4 waves, each owns 16 query rows
  hipLaunchKernelGGL(attn_fwd_f16wmma, grid, block, 0, stream, q, k, v, out);
}
